// AttenComm_62534723829940
// MI455X (gfx1250) — matrix operand order repaired and checked by reading, not compiled, not run on hardware
//
#include <hip/hip_runtime.h>
#include <hip/hip_bf16.h>

#define C_  256
#define H_  128
#define W_  256
#define HW_ (H_*W_)

typedef _Float16 v16h __attribute__((ext_vector_type(16)));
typedef float    v8f  __attribute__((ext_vector_type(8)));

union Frag { v16h h; unsigned u[8]; };

__device__ __forceinline__ unsigned pack2h(float a, float b) {
    _Float16 ha = (_Float16)a, hb = (_Float16)b;
    unsigned short ua, ub;
    __builtin_memcpy(&ua, &ha, 2);
    __builtin_memcpy(&ub, &hb, 2);
    return (unsigned)ua | ((unsigned)ub << 16);
}

__device__ __forceinline__ float diag_get(v8f d, int v) {
    float r = d[0];
    r = (v == 1) ? d[1] : r;
    r = (v == 2) ? d[2] : r;
    r = (v == 3) ? d[3] : r;
    r = (v == 4) ? d[4] : r;
    r = (v == 5) ? d[5] : r;
    r = (v == 6) ? d[6] : r;
    r = (v == 7) ? d[7] : r;
    return r;
}

__global__ __launch_bounds__(256)
void atten_comm_kernel(const float* __restrict__ x,
                       const float* __restrict__ tm,
                       float* __restrict__ out)
{
    // LDS: staging for WMMA fragments (rows padded to 17 dwords to dodge bank
    // conflicts) + per-pixel score accumulators. 41 KB total.
    __shared__ unsigned ldsA[256 * 17];
    __shared__ unsigned ldsB[256 * 17];
    __shared__ float    ldsDots[256 * 6];

    const int t  = threadIdx.x;          // pixel w within row
    const int b  = blockIdx.x >> 7;      // batch
    const int h  = blockIdx.x & 127;     // row
    const int w  = t;
    const int L  = t & 31;               // lane
    const int wv = t >> 5;               // wave id (8 waves, 32 pixels each)

    const int NN[4] = {5, 4, 3, 2};
    const int SS[4] = {0, 5, 9, 12};
    const int N = NN[b];
    const int s = SS[b];

    const int   rowoff = h * W_ + w;
    const float gx = 2.0f * (float)w * (1.0f / (float)(W_ - 1)) - 1.0f;
    const float gy = 2.0f * (float)h * (1.0f / (float)(H_ - 1)) - 1.0f;

    // ---- per-member bilinear precompute (members j = 1..4; j=0 is identity) ----
    int          off[4][4];
    float        wgt[4][4];
    const float* basep[4];
#pragma unroll
    for (int j = 1; j < 5; ++j) {
        const int ji = j - 1;
        basep[ji] = x + (size_t)(s + j) * C_ * HW_;
        if (j < N) {
            // t = M[b,0,j] rows (0,1), cols (0,1,3), with normalization scales
            const float* m  = tm + ((size_t)(b * 5 + 0) * 5 + j) * 16;
            const float a00 = m[0];
            const float a01 = m[1] * ((float)H_ / (float)W_);            // *0.5
            const float a02 = m[3] * (2.0f / (4.0f * 0.4f * (float)W_));
            const float a10 = m[4] * ((float)W_ / (float)H_);            // *2.0
            const float a11 = m[5];
            const float a12 = m[7] * (2.0f / (4.0f * 0.4f * (float)H_));
            const float cx = a00 * gx + a01 * gy + a02;
            const float cy = a10 * gx + a11 * gy + a12;
            const float px = (cx + 1.0f) * 0.5f * (float)(W_ - 1);
            const float py = (cy + 1.0f) * 0.5f * (float)(H_ - 1);
            const float fx0 = floorf(px), fy0 = floorf(py);
            const float wx = px - fx0, wy = py - fy0;
            const int   x0 = (int)fx0, y0 = (int)fy0;
#pragma unroll
            for (int k = 0; k < 4; ++k) {
                const int  dx = k & 1, dy = k >> 1;
                const int  xi = x0 + dx, yi = y0 + dy;
                const bool valid = (xi >= 0) && (xi <= W_ - 1) && (yi >= 0) && (yi <= H_ - 1);
                const int  xc = xi < 0 ? 0 : (xi > W_ - 1 ? W_ - 1 : xi);
                const int  yc = yi < 0 ? 0 : (yi > H_ - 1 ? H_ - 1 : yi);
                off[ji][k] = yc * W_ + xc;
                const float bw = (dx ? wx : 1.0f - wx) * (dy ? wy : 1.0f - wy);
                wgt[ji][k] = valid ? bw : 0.0f;
            }
        } else {
#pragma unroll
            for (int k = 0; k < 4; ++k) { off[ji][k] = 0; wgt[ji][k] = 0.0f; }
        }
    }

    const float* x0p = x + (size_t)s * C_ * HW_;   // ego member, exact copy path

    // ================= Phase 1: scores via WMMA (f16 in, f32 acc) =================
    // Per wave: 2 groups of 16 pixels; G = F0 * Fj^T per 32-channel chunk, keep diag.
    float acc0[5] = {0, 0, 0, 0, 0};
    float acc1[5] = {0, 0, 0, 0, 0};
    const bool own  = (L < 8) || (L >= 24);            // lanes that hold diagonal
    const int  vidx = (L < 8) ? L : (L - 24);          // which D VGPR holds it
    const int  ploc = (L < 8) ? L : (L - 16);          // which pixel (0..15) in tile

    for (int kc = 0; kc < 8; ++kc) {
        const int c0 = kc * 32;
        // stage F0 chunk: row t <- 32 channels of pixel t, packed f16 pairs
#pragma unroll 4
        for (int q = 0; q < 16; ++q) {
            const float fa = x0p[(c0 + 2 * q) * HW_ + rowoff];
            const float fb = x0p[(c0 + 2 * q + 1) * HW_ + rowoff];
            ldsA[t * 17 + q] = pack2h(fa, fb);
        }
        // build A fragments (16x32 f16 layout, ISA 7.12.2) for both 16-pixel groups
        Frag afr[2];
#pragma unroll
        for (int g = 0; g < 2; ++g) {
            const int prow = (wv * 32 + g * 16 + (L & 15)) * 17;
#pragma unroll
            for (int v = 0; v < 8; ++v) {
                const int kp = (L < 16) ? (v < 4 ? v : v + 4) : (v < 4 ? v + 4 : v + 8);
                afr[g].u[v] = ldsA[prow + kp];
            }
        }
#pragma unroll
        for (int j = 0; j < 5; ++j) {
            if (j < N) {                                  // uniform per block
                const unsigned* srcB = ldsA;              // j==0: F0 vs F0
                if (j > 0) {
                    const float* bp = basep[j - 1];
#pragma unroll 4
                    for (int q = 0; q < 16; ++q) {
                        const float* p0 = bp + (size_t)(c0 + 2 * q) * HW_;
                        const float* p1 = p0 + HW_;
                        const float fa = p0[off[j-1][0]] * wgt[j-1][0] + p0[off[j-1][1]] * wgt[j-1][1]
                                       + p0[off[j-1][2]] * wgt[j-1][2] + p0[off[j-1][3]] * wgt[j-1][3];
                        const float fb = p1[off[j-1][0]] * wgt[j-1][0] + p1[off[j-1][1]] * wgt[j-1][1]
                                       + p1[off[j-1][2]] * wgt[j-1][2] + p1[off[j-1][3]] * wgt[j-1][3];
                        ldsB[t * 17 + q] = pack2h(fa, fb);
                    }
                    srcB = ldsB;
                }
#pragma unroll
                for (int g = 0; g < 2; ++g) {
                    Frag bfr;
                    const int nrow = (wv * 32 + g * 16 + (L & 15)) * 17;
#pragma unroll
                    for (int v = 0; v < 8; ++v) {
                        const int kp = (L < 16) ? (v < 4 ? v : v + 4) : (v < 4 ? v + 4 : v + 8);   // B 32x16 f16 layout
                        bfr.u[v] = srcB[nrow + kp];
                    }
                    v8f dz = {};
                    v8f d = __builtin_amdgcn_wmma_f32_16x16x32_f16(
                        false, afr[g].h, false, bfr.h, (short)0, dz, false, false);
                    const float dv = diag_get(d, vidx);
                    if (own) { if (g == 0) acc0[j] += dv; else acc1[j] += dv; }
                }
            }
        }
    }

    if (own) {
#pragma unroll
        for (int j = 0; j < 5; ++j) {
            if (j < N) {
                ldsDots[(wv * 32 + ploc) * 6 + j]      = acc0[j];
                ldsDots[(wv * 32 + 16 + ploc) * 6 + j] = acc1[j];
            }
        }
    }
    __syncthreads();

    // softmax over members for pixel t (scores = dot / sqrt(256))
    float aw[5] = {0, 0, 0, 0, 0};
    {
        float sc[5];
        float mx = -1e30f;
#pragma unroll
        for (int j = 0; j < 5; ++j) {
            if (j < N) { sc[j] = ldsDots[t * 6 + j] * (1.0f / 16.0f); mx = fmaxf(mx, sc[j]); }
        }
        float sum = 0.0f;
#pragma unroll
        for (int j = 0; j < 5; ++j) {
            if (j < N) { sc[j] = __expf(sc[j] - mx); sum += sc[j]; }
        }
        const float inv = 1.0f / sum;
#pragma unroll
        for (int j = 0; j < 5; ++j) { if (j < N) aw[j] = sc[j] * inv; }
    }

    // ================= Phase 2: weighted combine (exact fp32) =================
    float* op = out + (size_t)b * C_ * HW_ + rowoff;
    for (int c = 0; c < C_; ++c) {
        float v = aw[0] * x0p[c * HW_ + rowoff];   // dominant ego term, exact
#pragma unroll
        for (int j = 1; j < 5; ++j) {
            if (j < N) {
                const float* p = basep[j - 1] + (size_t)c * HW_;
                v += aw[j] * (p[off[j-1][0]] * wgt[j-1][0] + p[off[j-1][1]] * wgt[j-1][1]
                            + p[off[j-1][2]] * wgt[j-1][2] + p[off[j-1][3]] * wgt[j-1][3]);
            }
        }
        op[(size_t)c * HW_] = v;
    }
}

extern "C" void kernel_launch(void* const* d_in, const int* in_sizes, int n_in,
                              void* d_out, int out_size, void* d_ws, size_t ws_size,
                              hipStream_t stream) {
    (void)in_sizes; (void)n_in; (void)out_size; (void)d_ws; (void)ws_size;
    const float* x  = (const float*)d_in[0];   // (14, 256, 128, 256) fp32
    // d_in[1] = rm (unused by reference forward), d_in[2] = record_len (constant)
    const float* tm = (const float*)d_in[3];   // (4, 5, 5, 4, 4) fp32
    float* out = (float*)d_out;                // (4, 256, 128, 256) fp32

    dim3 grid(4 * H_);     // one block per (batch, row)
    dim3 block(256);       // 8 waves, lane <-> w pixel (coalesced gathers)
    hipLaunchKernelGGL(atten_comm_kernel, grid, block, 0, stream, x, tm, out);
}
